// PerformerModel_55216099557515
// MI455X (gfx1250) — matrix operand order repaired and checked by reading, not compiled, not run on hardware
//
#include <hip/hip_runtime.h>
#include <hip/hip_bf16.h>

// Model dims
static constexpr int B_   = 16;
static constexpr int N_   = 8192;
static constexpr int IN_  = 64;
static constexpr int D_   = 128;
static constexpr int H_   = 8;
static constexpr int L_   = 3;
static constexpr int T_   = 424;
static constexpr long M_  = (long)B_ * N_;   // 131072 tokens

typedef __attribute__((ext_vector_type(16))) __bf16 v16bf;
typedef __attribute__((ext_vector_type(8)))  float  v8f;

union FragBF { v16bf v; unsigned int u[8]; };
union U128   { uint4 q; unsigned short s[8]; };

__device__ __forceinline__ unsigned short f2bf(float f) {
  unsigned int u = __float_as_uint(f);
  u += 0x7fffu + ((u >> 16) & 1u);           // round-to-nearest-even
  return (unsigned short)(u >> 16);
}
__device__ __forceinline__ float bf2f(unsigned short h) {
  return __uint_as_float(((unsigned int)h) << 16);
}

// ---- CDNA5 async global->LDS copy (ASYNCcnt path), GV addressing ----
// LDS offset taken from the vdst VGPR; global addr from 64-bit VGPR pair.
__device__ __forceinline__ void async_ld_b128(unsigned lds_off, const void* gaddr) {
  asm volatile("global_load_async_to_lds_b128 %0, %1, off"
               :: "v"(lds_off), "v"(gaddr) : "memory");
}
__device__ __forceinline__ void wait_async0() {
  asm volatile("s_wait_asynccnt 0x0" ::: "memory");
}

// ---------------------------------------------------------------- convert
__global__ __launch_bounds__(256)
void f2bf_kernel(const float* __restrict__ in, unsigned short* __restrict__ out, long n) {
  long i = (long)blockIdx.x * 256 + threadIdx.x;
  if (i < n) out[i] = f2bf(in[i]);
}

// ---------------------------------------------------------------- GEMM (bf16 WMMA)
// C[M,Nout] = A[M,K]@W[K,Nout]; block tile 128x64, 8 waves, wave tile 16x64.
// A-tile: async global->LDS (b128, ASYNCcnt). B-tile: VGPR + transposed ds_store.
// out_kind: 0 = bf16 store with activation (act 1: elu+1 for col<act_limit; act 2: exact gelu)
//           1 = f32 residual accumulate (f32_out += acc + bias)
//           2 = f32 store acc + bias + PE(row%N, col)   (embedding)
__global__ __launch_bounds__(256)
void gemm_bf16_wmma(const unsigned short* __restrict__ A,
                    const unsigned short* __restrict__ W,
                    const float* __restrict__ bias,
                    unsigned short* __restrict__ bf_out,
                    float* __restrict__ f32_out,
                    int K, int Nout, int out_kind, int act_mode, int act_limit)
{
  __shared__ unsigned short lA[128 * 32];   // [m][k]
  __shared__ unsigned short lB[64 * 32];    // [n][k]  (transposed)
  const int tid  = threadIdx.x;
  const int wave = tid >> 5, lane = tid & 31;
  const int hl   = lane >> 4, lm = lane & 15;
  const long r0  = (long)blockIdx.x * 128;
  const int  c0  = blockIdx.y * 64;

  // LDS byte offsets for this thread's two A slots (low 32 bits of generic ptr = LDS offset)
  unsigned aoff0, aoff1;
  {
    int slot0 = tid,       row0 = slot0 >> 2, seg0 = slot0 & 3;
    int slot1 = tid + 256, row1 = slot1 >> 2, seg1 = slot1 & 3;
    aoff0 = (unsigned)(size_t)&lA[row0 * 32 + seg0 * 8];
    aoff1 = (unsigned)(size_t)&lA[row1 * 32 + seg1 * 8];
  }

  v8f acc[4] = {};
  const int ksteps = K >> 5;

  for (int kt = 0; kt < ksteps; ++kt) {
    __syncthreads();   // prior-iteration fragment reads complete before refill
    // async A tile fill: 128 rows x 32 bf16, one b128 per slot
    {
      int row0 = tid >> 2,         seg0 = tid & 3;
      int row1 = (tid + 256) >> 2, seg1 = (tid + 256) & 3;
      async_ld_b128(aoff0, A + (r0 + row0) * K + kt * 32 + seg0 * 8);
      async_ld_b128(aoff1, A + (r0 + row1) * K + kt * 32 + seg1 * 8);
    }
    // cooperative B tile load (32 k-rows x 64 n-cols) -> LDS transposed [n][k]
    {
      int k = tid >> 3, seg = tid & 7;
      U128 w;
      w.q = *(const uint4*)(W + (long)(kt * 32 + k) * Nout + c0 + seg * 8);
      #pragma unroll
      for (int i = 0; i < 8; ++i) lB[(seg * 8 + i) * 32 + k] = w.s[i];
    }
    wait_async0();     // this wave's async LDS writes have landed
    __syncthreads();   // publish A (async) + B (ds stores) to all waves

    if (kt + 1 < ksteps)
      __builtin_prefetch(A + (r0 + wave * 16 + lm) * K + (kt + 1) * 32, 0, 1);

    // A fragment: 16x32 bf16, lane half hl holds k = hl*16 .. +15 contiguous
    FragBF a;
    const unsigned int* pa = (const unsigned int*)&lA[(wave * 16 + lm) * 32];
    #pragma unroll
    for (int v = 0; v < 8; ++v) {
      int k = hl * 16 + v * 2;
      a.u[v] = pa[k >> 1];
    }
    #pragma unroll
    for (int t = 0; t < 4; ++t) {
      // B fragment: lane n = t*16+lm, K = hl*16 .. +15 contiguous in LDS
      FragBF b;
      const unsigned int* pb = (const unsigned int*)&lB[(t * 16 + lm) * 32 + hl * 16];
      #pragma unroll
      for (int v = 0; v < 8; ++v) b.u[v] = pb[v];
      acc[t] = __builtin_amdgcn_wmma_f32_16x16x32_bf16(
          false, a.v, false, b.v, (short)0, acc[t], false, false);
    }
  }

  // epilogue: C/D layout — VGPR r: M = r + hl*8, N = lm
  #pragma unroll
  for (int t = 0; t < 4; ++t) {
    #pragma unroll
    for (int r = 0; r < 8; ++r) {
      long row = r0 + wave * 16 + (r + hl * 8);
      int  col = c0 + t * 16 + lm;
      float x = acc[t][r] + (bias ? bias[col] : 0.f);
      if (out_kind == 0) {
        if (act_mode == 1 && col < act_limit)
          x = (x > 0.f) ? (x + 1.f) : __expf(x);        // elu(x)+1
        else if (act_mode == 2)
          x = 0.5f * x * (1.f + erff(x * 0.70710678118654752f));  // exact gelu
        bf_out[row * Nout + col] = f2bf(x);
      } else if (out_kind == 1) {
        f32_out[row * Nout + col] += x;                 // residual add
      } else {
        long ntok = row & (N_ - 1);
        float freq = __expf(-9.210340371976184f * (float)(col & ~1) * (1.f / 128.f));
        float ang  = (float)ntok * freq;
        x += (col & 1) ? __cosf(ang) : __sinf(ang);     // positional encoding
        f32_out[row * Nout + col] = x;
      }
    }
  }
}

// ---------------------------------------------------------------- LayerNorm -> bf16
__global__ __launch_bounds__(256)
void ln_kernel(const float* __restrict__ h, const float* __restrict__ g,
               const float* __restrict__ be, unsigned short* __restrict__ out)
{
  long row = (long)blockIdx.x * 8 + (threadIdx.x >> 5);
  int lane = threadIdx.x & 31;
  const float* p = h + row * 128 + lane * 4;
  float x0 = p[0], x1 = p[1], x2 = p[2], x3 = p[3];
  float s = x0 + x1 + x2 + x3;
  #pragma unroll
  for (int o = 16; o; o >>= 1) s += __shfl_xor(s, o, 32);
  float m = s * (1.f / 128.f);
  float d0 = x0 - m, d1 = x1 - m, d2 = x2 - m, d3 = x3 - m;
  float v = d0 * d0 + d1 * d1 + d2 * d2 + d3 * d3;
  #pragma unroll
  for (int o = 16; o; o >>= 1) v += __shfl_xor(v, o, 32);
  float inv = rsqrtf(v * (1.f / 128.f) + 1e-5f);
  int c = lane * 4;
  unsigned short* q = out + row * 128 + c;
  q[0] = f2bf(d0 * inv * g[c + 0] + be[c + 0]);
  q[1] = f2bf(d1 * inv * g[c + 1] + be[c + 1]);
  q[2] = f2bf(d2 * inv * g[c + 2] + be[c + 2]);
  q[3] = f2bf(d3 * inv * g[c + 3] + be[c + 3]);
}

// ---------------------------------------------------------------- kv = k^T v per (b,h) via WMMA
// grid = B*H, block 256 (8 waves); wave w accumulates rows [w*1024, w*1024+1024)
__global__ __launch_bounds__(256)
void kv_kernel(const unsigned short* __restrict__ qkv,
               float* __restrict__ kvout, float* __restrict__ ksumout)
{
  int bh = blockIdx.x, bb = bh >> 3, hh = bh & 7;
  int tid = threadIdx.x, wave = tid >> 5, lane = tid & 31;
  int hl = lane >> 4, lm = lane & 15;
  __shared__ float skv[256];
  __shared__ float sks[16];
  if (tid < 256) skv[tid] = 0.f;
  if (tid < 16)  sks[tid] = 0.f;
  __syncthreads();

  const int ldq = 3 * D_;
  const long base = (long)bb * N_ * ldq;
  const int kcol = D_ + hh * 16 + lm;       // k section, d = lm  (A row)
  const int vcol = 2 * D_ + hh * 16 + lm;   // v section, e = lm  (B col)

  v8f acc = {};
  float ks = 0.f;
  for (int n0 = wave * 1024; n0 < wave * 1024 + 1024; n0 += 32) {
    FragBF a, bfr;
    #pragma unroll
    for (int v = 0; v < 8; ++v) {
      int ka = hl * 16 + v * 2;                               // A: K pair (= token offset)
      unsigned short x0 = qkv[base + (long)(n0 + ka)     * ldq + kcol];
      unsigned short x1 = qkv[base + (long)(n0 + ka + 1) * ldq + kcol];
      a.u[v] = (unsigned)x0 | ((unsigned)x1 << 16);
      ks += bf2f(x0) + bf2f(x1);                              // ksum partial (d = lm)
      int kb = hl * 16 + v * 2;                               // B: K pair
      unsigned short y0 = qkv[base + (long)(n0 + kb)     * ldq + vcol];
      unsigned short y1 = qkv[base + (long)(n0 + kb + 1) * ldq + vcol];
      bfr.u[v] = (unsigned)y0 | ((unsigned)y1 << 16);
    }
    acc = __builtin_amdgcn_wmma_f32_16x16x32_bf16(
        false, a.v, false, bfr.v, (short)0, acc, false, false);
  }
  #pragma unroll
  for (int r = 0; r < 8; ++r)
    atomicAdd(&skv[(r + hl * 8) * 16 + lm], acc[r]);
  atomicAdd(&sks[lm], ks);
  __syncthreads();
  if (tid < 256) kvout[bh * 256 + tid] = skv[tid];
  if (tid < 16)  ksumout[bh * 16 + tid] = sks[tid];
}

// ---------------------------------------------------------------- y = (q @ kv) / (q . ksum + eps)
__global__ __launch_bounds__(256)
void attn_out_kernel(const unsigned short* __restrict__ qkv,
                     const float* __restrict__ kv, const float* __restrict__ ksum,
                     unsigned short* __restrict__ y)
{
  long idx = (long)blockIdx.x * 256 + threadIdx.x;   // over M_*128
  int  c = (int)(idx & 127);
  long token = idx >> 7;
  int  bb = (int)(token >> 13);
  int  hh = c >> 4, e = c & 15;
  int  bh = bb * 8 + hh;
  const unsigned short* qp = qkv + token * (3 * D_) + hh * 16;  // q section
  const float* kvp = kv + bh * 256;
  const float* ksp = ksum + bh * 16;
  float num = 0.f, den = 0.f;
  #pragma unroll
  for (int d = 0; d < 16; ++d) {
    float q = bf2f(qp[d]);
    num += q * kvp[d * 16 + e];
    den += q * ksp[d];
  }
  y[token * D_ + c] = f2bf(num / (den + 1e-6f));
}

// ---------------------------------------------------------------- mean pool over sequence
__global__ __launch_bounds__(256)
void pool_kernel(const float* __restrict__ h, float* __restrict__ pooled)
{
  int b = blockIdx.x, tid = threadIdx.x;
  int d = tid & 127, hf = tid >> 7;
  const float* p = h + ((long)b * N_ + (long)hf * 4096) * D_ + d;
  float s = 0.f;
  for (int n = 0; n < 4096; ++n) s += p[(long)n * D_];
  __shared__ float ls[256];
  ls[tid] = s;
  __syncthreads();
  if (tid < 128) pooled[b * D_ + tid] = (ls[tid] + ls[tid + 128]) * (1.f / (float)N_);
}

// ---------------------------------------------------------------- final LN + MLP head
__global__ __launch_bounds__(256)
void head_kernel(const float* __restrict__ pooled, const float* __restrict__ g,
                 const float* __restrict__ be,
                 const float* __restrict__ w1, const float* __restrict__ b1,
                 const float* __restrict__ w2, const float* __restrict__ b2,
                 float* __restrict__ out)
{
  int b = blockIdx.x, tid = threadIdx.x;
  __shared__ float tmp[256];
  __shared__ float xr[128];
  __shared__ float hid[64];
  float val = (tid < 128) ? pooled[b * D_ + tid] : 0.f;
  tmp[tid] = val;
  __syncthreads();
  for (int s = 128; s >= 1; s >>= 1) { if (tid < s) tmp[tid] += tmp[tid + s]; __syncthreads(); }
  float mean = tmp[0] * (1.f / 128.f);
  __syncthreads();
  float d = val - mean;
  tmp[tid] = (tid < 128) ? d * d : 0.f;
  __syncthreads();
  for (int s = 128; s >= 1; s >>= 1) { if (tid < s) tmp[tid] += tmp[tid + s]; __syncthreads(); }
  float inv = rsqrtf(tmp[0] * (1.f / 128.f) + 1e-5f);
  __syncthreads();
  if (tid < 128) xr[tid] = d * inv * g[tid] + be[tid];
  __syncthreads();
  if (tid < 64) {
    float s = b1[tid];
    for (int k = 0; k < 128; ++k) s += xr[k] * w1[k * 64 + tid];
    hid[tid] = fmaxf(s, 0.f);
  }
  __syncthreads();
  for (int t = tid; t < T_; t += 256) {
    float s = b2[t];
    for (int k = 0; k < 64; ++k) s += hid[k] * w2[k * T_ + t];
    out[b * T_ + t] = s;
  }
}

// ---------------------------------------------------------------- launcher
extern "C" void kernel_launch(void* const* d_in, const int* in_sizes, int n_in,
                              void* d_out, int out_size, void* d_ws, size_t ws_size,
                              hipStream_t stream)
{
  (void)in_sizes; (void)n_in; (void)out_size; (void)ws_size;
  const float* x      = (const float*)d_in[0];
  const float* w_in   = (const float*)d_in[1];
  const float* b_in   = (const float*)d_in[2];
  const float* ln1_g  = (const float*)d_in[3];
  const float* ln1_b  = (const float*)d_in[4];
  const float* w_qkv  = (const float*)d_in[5];
  const float* w_proj = (const float*)d_in[6];
  const float* b_proj = (const float*)d_in[7];
  const float* ln2_g  = (const float*)d_in[8];
  const float* ln2_b  = (const float*)d_in[9];
  const float* w_ff1  = (const float*)d_in[10];
  const float* b_ff1  = (const float*)d_in[11];
  const float* w_ff2  = (const float*)d_in[12];
  const float* b_ff2  = (const float*)d_in[13];
  const float* ln_g   = (const float*)d_in[14];
  const float* ln_b   = (const float*)d_in[15];
  const float* w_o1   = (const float*)d_in[16];
  const float* b_o1   = (const float*)d_in[17];
  const float* w_o2   = (const float*)d_in[18];
  const float* b_o2   = (const float*)d_in[19];

  char* ws = (char*)d_ws;
  size_t off = 0;
  auto alloc = [&](size_t bytes) -> void* {
    void* p = ws + off;
    off += (bytes + 255) & ~(size_t)255;
    return p;
  };
  float*          h        = (float*)         alloc(M_ * D_ * 4);
  unsigned short* xn       = (unsigned short*)alloc(M_ * D_ * 2);
  unsigned short* qkv      = (unsigned short*)alloc(M_ * 3 * D_ * 2);
  unsigned short* y        = (unsigned short*)alloc(M_ * D_ * 2);
  unsigned short* ff       = (unsigned short*)alloc(M_ * 4 * D_ * 2);
  unsigned short* xbf      = (unsigned short*)alloc(M_ * IN_ * 2);
  unsigned short* w_in_bf  = (unsigned short*)alloc((size_t)IN_ * D_ * 2);
  unsigned short* w_qkv_bf = (unsigned short*)alloc((size_t)L_ * D_ * 3 * D_ * 2);
  unsigned short* w_prj_bf = (unsigned short*)alloc((size_t)L_ * D_ * D_ * 2);
  unsigned short* w_ff1_bf = (unsigned short*)alloc((size_t)L_ * D_ * 4 * D_ * 2);
  unsigned short* w_ff2_bf = (unsigned short*)alloc((size_t)L_ * 4 * D_ * D_ * 2);
  float*          kv       = (float*)         alloc((size_t)B_ * H_ * 256 * 4);
  float*          ksum     = (float*)         alloc((size_t)B_ * H_ * 16 * 4);
  float*          pooled   = (float*)         alloc((size_t)B_ * D_ * 4);

  auto cv = [&](const float* src, unsigned short* dst, long n) {
    f2bf_kernel<<<(unsigned)((n + 255) / 256), 256, 0, stream>>>(src, dst, n);
  };
  cv(x,      xbf,      M_ * IN_);
  cv(w_in,   w_in_bf,  (long)IN_ * D_);
  cv(w_qkv,  w_qkv_bf, (long)L_ * D_ * 3 * D_);
  cv(w_proj, w_prj_bf, (long)L_ * D_ * D_);
  cv(w_ff1,  w_ff1_bf, (long)L_ * D_ * 4 * D_);
  cv(w_ff2,  w_ff2_bf, (long)L_ * 4 * D_ * D_);

  const unsigned gM = (unsigned)(M_ / 128);

  // h = x @ w_in + b_in + PE
  gemm_bf16_wmma<<<dim3(gM, 2), 256, 0, stream>>>(
      xbf, w_in_bf, b_in, nullptr, h, IN_, D_, /*kind*/2, 0, 0);

  for (int l = 0; l < L_; ++l) {
    // attention
    ln_kernel<<<(unsigned)(M_ / 8), 256, 0, stream>>>(h, ln1_g + l * D_, ln1_b + l * D_, xn);
    gemm_bf16_wmma<<<dim3(gM, 6), 256, 0, stream>>>(
        xn, w_qkv_bf + (size_t)l * D_ * 3 * D_, nullptr, qkv, nullptr,
        D_, 3 * D_, /*kind*/0, /*act*/1, /*limit*/2 * D_);   // elu+1 on q,k cols
    kv_kernel<<<B_ * H_, 256, 0, stream>>>(qkv, kv, ksum);
    attn_out_kernel<<<(unsigned)(M_ * D_ / 256), 256, 0, stream>>>(qkv, kv, ksum, y);
    gemm_bf16_wmma<<<dim3(gM, 2), 256, 0, stream>>>(
        y, w_prj_bf + (size_t)l * D_ * D_, b_proj + l * D_, nullptr, h,
        D_, D_, /*kind*/1, 0, 0);                            // h += proj

    // feed-forward
    ln_kernel<<<(unsigned)(M_ / 8), 256, 0, stream>>>(h, ln2_g + l * D_, ln2_b + l * D_, xn);
    gemm_bf16_wmma<<<dim3(gM, 8), 256, 0, stream>>>(
        xn, w_ff1_bf + (size_t)l * D_ * 4 * D_, b_ff1 + l * 4 * D_, ff, nullptr,
        D_, 4 * D_, /*kind*/0, /*act*/2, 0);                 // gelu
    gemm_bf16_wmma<<<dim3(gM, 2), 256, 0, stream>>>(
        ff, w_ff2_bf + (size_t)l * 4 * D_ * D_, b_ff2 + l * D_, nullptr, h,
        4 * D_, D_, /*kind*/1, 0, 0);                        // h += ff2
  }

  pool_kernel<<<B_, 256, 0, stream>>>(h, pooled);
  head_kernel<<<B_, 256, 0, stream>>>(pooled, ln_g, ln_b, w_o1, b_o1, w_o2, b_o2,
                                      (float*)d_out);
}
